// PhysicsRULModel_51513837748586
// MI455X (gfx1250) — matrix operand order repaired and checked by reading, not compiled, not run on hardware
//
#include <hip/hip_runtime.h>
#include <math.h>

typedef __attribute__((ext_vector_type(16))) _Float16 v16h;
typedef __attribute__((ext_vector_type(8)))  float    v8f;

#define B_    256
#define T_    2048
#define H_    64
#define G_    256   // 4*H
#define BT_   16    // batch tile (WMMA M)
#define NT_   512   // threads per block = 16 wave32
#define RGAS  8.314f

__device__ __forceinline__ v8f wmma16(v16h a, v16h b, v8f c) {
  // D = A(16x32,f16) * B(32x16,f16) + C(16x16,f32)
  return __builtin_amdgcn_wmma_f32_16x16x32_f16(false, a, false, b, (short)0, c, false, false);
}

// A-fragment (16x32 f16) from LDS hidden-state buffer hb[16][64], K offset koff.
// Contiguous K order (this code's choice): lanes0-15 M=0..15; halves 0..15 -> K=0..15;
// lanes16-31 add +16 to K (matches B fragment).
__device__ __forceinline__ v16h load_afrag(const _Float16* hb, int m, int koff, int lgrp) {
  const _Float16* row = hb + m * H_;
  v16h a;
#pragma unroll
  for (int i = 0; i < 16; ++i) {
    int k = koff + i + lgrp * 16;
    a[i] = row[k];
  }
  return a;
}

// branch-free activations on the hardware fast path:
// v_exp_f32 (TRANS) + v_add + raw v_rcp_f32 (no IEEE div expansion).
// Saturation is exact: exp->inf => rcp->0 => sigm->0/tanh->1; exp->0 => sigm->1/tanh->-1.
__device__ __forceinline__ float sigm(float x) {
  return __builtin_amdgcn_rcpf(1.0f + __expf(-x));
}
__device__ __forceinline__ float tanh_fast(float x) {
  return 1.0f - 2.0f * __builtin_amdgcn_rcpf(1.0f + __expf(2.0f * x));
}

__global__ __launch_bounds__(NT_, 1)
void fused_physics_rul(
    const float* __restrict__ ts,   const float* __restrict__ stat,
    const float* __restrict__ Wih0, const float* __restrict__ Whh0,
    const float* __restrict__ bih0, const float* __restrict__ bhh0,
    const float* __restrict__ Wih1, const float* __restrict__ Whh1,
    const float* __restrict__ bih1, const float* __restrict__ bhh1,
    const float* __restrict__ seW1, const float* __restrict__ seb1,
    const float* __restrict__ seW2, const float* __restrict__ seb2,
    const float* __restrict__ pbW1, const float* __restrict__ pbb1,
    const float* __restrict__ pbW2, const float* __restrict__ pbb2,
    const float* __restrict__ pbW3, const float* __restrict__ pbb3,
    const float* __restrict__ drW1, const float* __restrict__ drb1,
    const float* __restrict__ drW2, const float* __restrict__ drb2,
    const float* __restrict__ drW3, const float* __restrict__ drb3,
    const float* __restrict__ ruW1, const float* __restrict__ rub1,
    const float* __restrict__ ruW2, const float* __restrict__ rub2,
    float* __restrict__ out)
{
  __shared__ float    gates[BT_ * G_];          // 16KB; reused as head scratch
  __shared__ _Float16 h1h[BT_ * H_];            // layer0 hidden (f16, WMMA A)
  __shared__ _Float16 h2h[BT_ * H_];            // layer1 hidden (f16, WMMA A + heads)
  __shared__ float    bias0[G_], bias1[G_];
  __shared__ float    xb[2][BT_][2];            // double-buffered x_t
  __shared__ float    sst[BT_ * 32];            // system states
  __shared__ float    EaV[BT_], EaJ[BT_], lgA[BT_], expA[BT_], hprev[BT_];
  __shared__ float    l_seW1[32*64], l_seb1[32], l_seW2[32*32], l_seb2[32];
  __shared__ float    l_drW1[64*35], l_drb1[64], l_drW2[32*64], l_drb2[32], l_drW3[32];
  __shared__ float    l_ruW1[32*33], l_rub1[32], l_ruW2[32];
  __shared__ float    l_drb3, l_rub2;

  const int tid  = threadIdx.x;
  const int lane = tid & 31;
  const int wav  = tid >> 5;
  const int bg0  = blockIdx.x * BT_;

  // ---------------- one-time init ----------------
  {
    auto cp = [&](float* d, const float* s, int n) {
      for (int i = tid; i < n; i += NT_) d[i] = s[i];
    };
    cp(l_seW1, seW1, 32*64); cp(l_seb1, seb1, 32);
    cp(l_seW2, seW2, 32*32); cp(l_seb2, seb2, 32);
    cp(l_drW1, drW1, 64*35); cp(l_drb1, drb1, 64);
    cp(l_drW2, drW2, 32*64); cp(l_drb2, drb2, 32);
    cp(l_drW3, drW3, 32);
    cp(l_ruW1, ruW1, 32*33); cp(l_rub1, rub1, 32);
    cp(l_ruW2, ruW2, 32);
    if (tid == 0) { l_drb3 = drb3[0]; l_rub2 = rub2[0]; }
    for (int n = tid; n < G_; n += NT_) { bias0[n] = bih0[n] + bhh0[n]; bias1[n] = bih1[n] + bhh1[n]; }
    for (int u = tid; u < BT_ * H_; u += NT_) { h1h[u] = (_Float16)0.0f; h2h[u] = (_Float16)0.0f; }
    if (tid < 32) {                              // x_0 into buffer 0
      const int b = tid >> 1, d = tid & 1;
      xb[0][b][d] = ts[((size_t)(bg0 + b) * T_) * 2 + d];
    }
  }

  // physics branch (per batch row, time invariant): 3 -> 32 -> 16 -> 2
  if (tid < BT_) {
    const int bg = bg0 + tid;
    const float s0 = stat[bg*3+0], s1v = stat[bg*3+1], s2 = stat[bg*3+2];
    float p1[32];
#pragma unroll
    for (int j = 0; j < 32; ++j)
      p1[j] = fmaxf(pbb1[j] + pbW1[j*3+0]*s0 + pbW1[j*3+1]*s1v + pbW1[j*3+2]*s2, 0.0f);
    float p2[16];
#pragma unroll
    for (int j = 0; j < 16; ++j) {
      float a = pbb2[j];
#pragma unroll
      for (int k = 0; k < 32; ++k) a += pbW2[j*32+k]*p1[k];
      p2[j] = fmaxf(a, 0.0f);
    }
    float Ea = pbb3[0], lA = pbb3[1];
#pragma unroll
    for (int k = 0; k < 16; ++k) { Ea += pbW3[k]*p2[k]; lA += pbW3[16+k]*p2[k]; }
    EaV[tid] = Ea; EaJ[tid] = Ea*1000.0f; lgA[tid] = lA; expA[tid] = __expf(lA);
    hprev[tid] = 1.0f;
  }
  __syncthreads();

  // per-wave B-matrix fragments (f32 global -> f16 regs), N tile = wave id
  const int nG   = (wav << 4) + (lane & 15);   // gate column 0..255
  const int lgrp = (lane >> 4) & 1;
  const int kgrp = lgrp << 4;                  // B layout: lanes0-15 K=0..15, lanes16-31 K=16..31
  v16h bWhh0[2], bWih1[2], bWhh1[2];
#pragma unroll
  for (int c = 0; c < 2; ++c) {
    const int kb = kgrp + 32*c;
#pragma unroll
    for (int i = 0; i < 16; ++i) {
      bWhh0[c][i] = (_Float16)Whh0[nG*H_ + kb + i];
      bWih1[c][i] = (_Float16)Wih1[nG*H_ + kb + i];
      bWhh1[c][i] = (_Float16)Whh1[nG*H_ + kb + i];
    }
  }
  const float w00 = Wih0[nG*2+0], w01 = Wih0[nG*2+1];
  const float bb0 = bias0[nG],    bb1 = bias1[nG];
  const int   mlo = lgrp << 3;                 // accum: M = mlo + r, N = lane&15

  // fixed per-thread ownership of 2 LSTM units; cell state stays in registers
  const int u0 = tid*2, u1 = tid*2 + 1;
  const int cb0 = u0 >> 6, cj0 = u0 & 63;
  const int cb1 = u1 >> 6, cj1 = u1 & 63;
  float c1x = 0.f, c1y = 0.f, c2x = 0.f, c2y = 0.f;

  float* scr = gates;
  const size_t BTsz = (size_t)B_ * T_;

  for (int t = 0; t < T_; ++t) {
    const int cur = t & 1, nxt = cur ^ 1;
    __syncthreads();                            // protect xb[nxt] / scr reuse across steps

    // stream x_{t+1} into the alternate buffer while this step computes;
    // prefetch x_{t+2} toward L2/WGP$ to keep HBM off the critical path.
    if (tid < 32 && t + 1 < T_) {
      const int b = tid >> 1, d = tid & 1;
      xb[nxt][b][d] = ts[((size_t)(bg0 + b) * T_ + t + 1) * 2 + d];
      if (t + 2 < T_)
        __builtin_prefetch(&ts[((size_t)(bg0 + b) * T_ + t + 2) * 2 + d], 0, 1);
    }
    const float (*xc)[2] = xb[cur];

    // ---- LSTM layer 0: gates = bias + x@Wih0^T (VALU, K=2) + h1@Whh0^T (WMMA)
    {
      v8f acc;
#pragma unroll
      for (int r = 0; r < 8; ++r)
        acc[r] = bb0 + xc[mlo + r][0]*w00 + xc[mlo + r][1]*w01;
      v16h a0 = load_afrag(h1h, lane & 15, 0,  lgrp);
      v16h a1 = load_afrag(h1h, lane & 15, 32, lgrp);
      acc = wmma16(a0, bWhh0[0], acc);
      acc = wmma16(a1, bWhh0[1], acc);
#pragma unroll
      for (int r = 0; r < 8; ++r) gates[(mlo + r) * G_ + nG] = acc[r];
    }
    __syncthreads();

    // ---- layer 0 cell update (branch-free activations)
    {
      float iv = gates[cb0*G_ + cj0],       fv = gates[cb0*G_ + cj0 + 64];
      float gv = gates[cb0*G_ + cj0 + 128], ov = gates[cb0*G_ + cj0 + 192];
      c1x = sigm(fv)*c1x + sigm(iv)*tanh_fast(gv);
      h1h[cb0*H_ + cj0] = (_Float16)(sigm(ov)*tanh_fast(c1x));
      iv = gates[cb1*G_ + cj1];       fv = gates[cb1*G_ + cj1 + 64];
      gv = gates[cb1*G_ + cj1 + 128]; ov = gates[cb1*G_ + cj1 + 192];
      c1y = sigm(fv)*c1y + sigm(iv)*tanh_fast(gv);
      h1h[cb1*H_ + cj1] = (_Float16)(sigm(ov)*tanh_fast(c1y));
    }
    __syncthreads();

    // ---- LSTM layer 1: gates = bias + h1@Wih1^T + h2@Whh1^T (4 WMMAs)
    {
      v8f acc;
#pragma unroll
      for (int r = 0; r < 8; ++r) acc[r] = bb1;
      v16h a0 = load_afrag(h1h, lane & 15, 0,  lgrp);
      v16h a1 = load_afrag(h1h, lane & 15, 32, lgrp);
      v16h a2 = load_afrag(h2h, lane & 15, 0,  lgrp);
      v16h a3 = load_afrag(h2h, lane & 15, 32, lgrp);
      acc = wmma16(a0, bWih1[0], acc);
      acc = wmma16(a1, bWih1[1], acc);
      acc = wmma16(a2, bWhh1[0], acc);
      acc = wmma16(a3, bWhh1[1], acc);
#pragma unroll
      for (int r = 0; r < 8; ++r) gates[(mlo + r) * G_ + nG] = acc[r];
    }
    __syncthreads();

    // ---- layer 1 cell update
    {
      float iv = gates[cb0*G_ + cj0],       fv = gates[cb0*G_ + cj0 + 64];
      float gv = gates[cb0*G_ + cj0 + 128], ov = gates[cb0*G_ + cj0 + 192];
      c2x = sigm(fv)*c2x + sigm(iv)*tanh_fast(gv);
      h2h[cb0*H_ + cj0] = (_Float16)(sigm(ov)*tanh_fast(c2x));
      iv = gates[cb1*G_ + cj1];       fv = gates[cb1*G_ + cj1 + 64];
      gv = gates[cb1*G_ + cj1 + 128]; ov = gates[cb1*G_ + cj1 + 192];
      c2y = sigm(fv)*c2y + sigm(iv)*tanh_fast(gv);
      h2h[cb1*H_ + cj1] = (_Float16)(sigm(ov)*tanh_fast(c2y));
    }
    __syncthreads();

    // ---- system encoder: relu(seW1@h2+b1) -> seW2@.+b2   (512 outputs each)
    {
      const int b = tid >> 5, j = tid & 31;
      float a = l_seb1[j];
#pragma unroll
      for (int k = 0; k < H_; ++k) a += l_seW1[j*H_ + k] * (float)h2h[b*H_ + k];
      scr[b*32 + j] = fmaxf(a, 0.0f);
    }
    __syncthreads();
    {
      const int b = tid >> 5, j = tid & 31;
      float a = l_seb2[j];
#pragma unroll
      for (int k = 0; k < 32; ++k) a += l_seW2[j*32 + k] * scr[b*32 + k];
      sst[b*32 + j] = a;
    }
    __syncthreads();

    // ---- degradation MLP: [sst(32), T, Ea, logA] -> 64 relu
    {
#pragma unroll
      for (int q = 0; q < 2; ++q) {
        const int u = tid*2 + q, b = u >> 6, j = u & 63;
        float a = l_drb1[j];
#pragma unroll
        for (int k = 0; k < 32; ++k) a += l_drW1[j*35 + k] * sst[b*32 + k];
        a += l_drW1[j*35 + 32] * xc[b][0];
        a += l_drW1[j*35 + 33] * EaV[b];
        a += l_drW1[j*35 + 34] * lgA[b];
        scr[u] = fmaxf(a, 0.0f);                // scr[0..1023]
      }
    }
    __syncthreads();
    {
      const int b = tid >> 5, j = tid & 31;     // 64 -> 32 relu
      float a = l_drb2[j];
#pragma unroll
      for (int k = 0; k < 64; ++k) a += l_drW2[j*64 + k] * scr[b*64 + k];
      scr[1024 + b*32 + j] = fmaxf(a, 0.0f);
    }
    __syncthreads();

    // ---- deg rate + Arrhenius + health scan + stores (16 lanes)
    if (tid < BT_) {
      const int b = tid;
      float deg = l_drb3;
#pragma unroll
      for (int k = 0; k < 32; ++k) deg += l_drW3[k] * scr[1024 + b*32 + k];
      const float tK   = xc[b][0] + 273.15f;
      const float arr  = expA[b] * __expf(-EaJ[b] * __builtin_amdgcn_rcpf(RGAS * tK));
      const float comb = 0.7f*deg + 0.3f*arr;
      float hn = hprev[b] - comb;
      hn = fminf(fmaxf(hn, 0.0f), 1.0f);
      hprev[b] = hn;
      out[BTsz   + (size_t)(bg0 + b)*T_ + t] = deg;   // degradation_rates
      out[2*BTsz + (size_t)(bg0 + b)*T_ + t] = hn;    // health
    }
    __syncthreads();

    // ---- RUL head: [sst(32), health] -> 32 relu -> 1
    {
      const int b = tid >> 5, j = tid & 31;
      float a = l_rub1[j];
#pragma unroll
      for (int k = 0; k < 32; ++k) a += l_ruW1[j*33 + k] * sst[b*32 + k];
      a += l_ruW1[j*33 + 32] * hprev[b];
      scr[1536 + b*32 + j] = fmaxf(a, 0.0f);
    }
    __syncthreads();
    if (tid < BT_) {
      const int b = tid;
      float a = l_rub2;
#pragma unroll
      for (int k = 0; k < 32; ++k) a += l_ruW2[k] * scr[1536 + b*32 + k];
      out[(size_t)(bg0 + b)*T_ + t] = fmaxf(a, 0.0f); // rul
    }
  }
}

extern "C" void kernel_launch(void* const* d_in, const int* in_sizes, int n_in,
                              void* d_out, int out_size, void* d_ws, size_t ws_size,
                              hipStream_t stream) {
  (void)in_sizes; (void)n_in; (void)out_size; (void)d_ws; (void)ws_size;
  const float* p[30];
  for (int i = 0; i < 30; ++i) p[i] = (const float*)d_in[i];
  fused_physics_rul<<<dim3(B_ / BT_), dim3(NT_), 0, stream>>>(
      p[0],  p[1],  p[2],  p[3],  p[4],  p[5],  p[6],  p[7],  p[8],  p[9],
      p[10], p[11], p[12], p[13], p[14], p[15], p[16], p[17], p[18], p[19],
      p[20], p[21], p[22], p[23], p[24], p[25], p[26], p[27], p[28], p[29],
      (float*)d_out);
}
